// DecoderBlock_27891517620396
// MI455X (gfx1250) — matrix operand order repaired and checked by reading, not compiled, not run on hardware
//
#include <hip/hip_runtime.h>

// ---------------------------------------------------------------------------
// CDNA5 (gfx1250) transformer decoder block using v_wmma_f32_16x16x32_bf16.
// wave32; WMMA fragment layouts per CDNA5 ISA 7.12.2.
// GEMMs: 128x128 block tile, K-step 64, double-buffered LDS + register
// prefetch (one barrier per K-step, 16 WMMAs per wave per step).
// Attention: double-buffered LDS KV tiles + WGP-scope global_prefetch_b8
// (no long-lived prefetch registers -> no scratch spills).
// ---------------------------------------------------------------------------

typedef __attribute__((ext_vector_type(16))) __bf16 v16bf;
typedef __attribute__((ext_vector_type(8)))  float  v8f;

#define DEVI __device__ __forceinline__

DEVI __bf16 f2bf(float f) { return (__bf16)f; }

union Pack16 { uint4 q[2]; __bf16 h[16]; };
union FragBF { v16bf v; uint4 q[2]; };

DEVI v8f wmma_bf16(const FragBF& a, const FragBF& b, v8f c) {
    return __builtin_amdgcn_wmma_f32_16x16x32_bf16(
        false, a.v, false, b.v, (short)0, c, false, false);
}

// ---------------------------------------------------------------------------
// fp32 -> bf16 conversion (weights), 4 elems per thread
// ---------------------------------------------------------------------------
__global__ __launch_bounds__(256)
void cvt_f32_bf16(const float* __restrict__ in, __bf16* __restrict__ out, int n4) {
    int i = blockIdx.x * 256 + threadIdx.x;
    if (i >= n4) return;
    float4 v = ((const float4*)in)[i];
    union { __bf16 h[4]; uint2 u; } p;
    p.h[0] = f2bf(v.x); p.h[1] = f2bf(v.y); p.h[2] = f2bf(v.z); p.h[3] = f2bf(v.w);
    ((uint2*)out)[i] = p.u;
}

// ---------------------------------------------------------------------------
// LayerNorm over d=1024, one row per 256-thread block, bf16 output
// ---------------------------------------------------------------------------
__global__ __launch_bounds__(256)
void ln_bf16(const float* __restrict__ x, const float* __restrict__ g,
             const float* __restrict__ bt, __bf16* __restrict__ y) {
    const int row = blockIdx.x, t = threadIdx.x;
    const float4 v = ((const float4*)(x + (size_t)row * 1024))[t];
    float s  = v.x + v.y + v.z + v.w;
    float s2 = v.x * v.x + v.y * v.y + v.z * v.z + v.w * v.w;
#pragma unroll
    for (int off = 16; off; off >>= 1) {
        s  += __shfl_xor(s,  off);
        s2 += __shfl_xor(s2, off);
    }
    __shared__ float sh[16];
    const int wv = t >> 5, lane = t & 31;
    if (lane == 0) { sh[wv] = s; sh[8 + wv] = s2; }
    __syncthreads();
    if (t == 0) {
        float a = 0.f, b2 = 0.f;
        for (int i = 0; i < 8; ++i) { a += sh[i]; b2 += sh[8 + i]; }
        sh[0] = a; sh[8] = b2;
    }
    __syncthreads();
    const float mu   = sh[0] * (1.0f / 1024.0f);
    const float var  = sh[8] * (1.0f / 1024.0f) - mu * mu;
    const float rstd = rsqrtf(var + 1e-5f);
    const float4 gg = ((const float4*)g)[t];
    const float4 bb = ((const float4*)bt)[t];
    __bf16* py = y + (size_t)row * 1024 + t * 4;
    py[0] = f2bf((v.x - mu) * rstd * gg.x + bb.x);
    py[1] = f2bf((v.y - mu) * rstd * gg.y + bb.y);
    py[2] = f2bf((v.z - mu) * rstd * gg.z + bb.z);
    py[3] = f2bf((v.w - mu) * rstd * gg.w + bb.w);
}

// ---------------------------------------------------------------------------
// Tiled WMMA GEMM: C[M,N] = A[M,K](bf16) * B[K,N](bf16) + bias, epilogue:
//   EPI=0: bf16 out ; EPI=1: bf16 GELU(out) ; EPI=2: f32 out + residual
// 256 threads = 8 waves (4 along M x 2 along N), wave tile 32x64.
// K-step 64, double-buffered LDS, register prefetch, WGP-scope prefetch.
// ---------------------------------------------------------------------------
template <int EPI>
__global__ __launch_bounds__(256)
void gemm_bf16_wmma(const __bf16* __restrict__ A, const __bf16* __restrict__ Bw,
                    const float* __restrict__ bias, const float* __restrict__ resid,
                    void* __restrict__ outp, int M, int N, int K) {
    __shared__ __align__(16) __bf16 As[2][128][72];  // row-major [m][k], 144B stride
    __shared__ __align__(16) __bf16 Bs[2][128][72];  // transposed [n][k]

    const int t = threadIdx.x;
    const int lane = t & 31, wv = t >> 5;
    const int lh = lane >> 4, ln16 = lane & 15;
    const int wm = wv & 3, wn = wv >> 2;
    const int bm = blockIdx.y * 128, bn = blockIdx.x * 128;

    v8f acc[2][4];
#pragma unroll
    for (int mt = 0; mt < 2; ++mt)
#pragma unroll
        for (int nt = 0; nt < 4; ++nt)
#pragma unroll
            for (int r = 0; r < 8; ++r) acc[mt][nt][r] = 0.0f;

    const int arow = t >> 1, acb = (t & 1) * 32;      // A: 32 elems/thread
    const int bk = t & 63, bnb = (t >> 6) * 32;       // B: 32 elems/thread

    uint4  ra[4];
    Pack16 rb[2];

    auto gload = [&](int k0) {
        const uint4* ga = (const uint4*)(A + (size_t)(bm + arow) * K + k0 + acb);
        ra[0] = ga[0]; ra[1] = ga[1]; ra[2] = ga[2]; ra[3] = ga[3];
        const uint4* gb = (const uint4*)(Bw + (size_t)(k0 + bk) * N + bn + bnb);
        rb[0].q[0] = gb[0]; rb[0].q[1] = gb[1];
        rb[1].q[0] = gb[2]; rb[1].q[1] = gb[3];
    };
    auto sstore = [&](int buf) {
        uint4* pa = (uint4*)&As[buf][arow][acb];
        pa[0] = ra[0]; pa[1] = ra[1]; pa[2] = ra[2]; pa[3] = ra[3];
#pragma unroll
        for (int i = 0; i < 16; ++i) Bs[buf][bnb + i][bk] = rb[0].h[i];
#pragma unroll
        for (int i = 0; i < 16; ++i) Bs[buf][bnb + 16 + i][bk] = rb[1].h[i];
    };

    const int nk = K >> 6;
    gload(0);
    sstore(0);
    __syncthreads();

    for (int kt = 0; kt < nk; ++kt) {
        if (kt + 1 < nk) gload((kt + 1) << 6);          // overlap with WMMAs
        if (kt + 2 < nk) {                              // WGP-scope prefetch hints
            __builtin_prefetch(A + (size_t)(bm + arow) * K + ((kt + 2) << 6) + acb, 0, 3);
            __builtin_prefetch(Bw + (size_t)(((kt + 2) << 6) + bk) * N + bn + bnb, 0, 3);
        }
        const int buf = kt & 1;

#pragma unroll
        for (int kk = 0; kk < 2; ++kk) {  // per-half fragments: lower VGPR pressure
            FragBF fa[2], fb[4];
#pragma unroll
            for (int mt = 0; mt < 2; ++mt) {  // A layout: K {0..7,16..23}/{8..15,24..31}
                const __bf16* p = &As[buf][wm * 32 + mt * 16 + ln16][kk * 32 + lh * 8];
                fa[mt].q[0] = *(const uint4*)p;
                fa[mt].q[1] = *(const uint4*)(p + 16);
            }
#pragma unroll
            for (int nt = 0; nt < 4; ++nt) {  // B layout: K {0..7,16..23}/{8..15,24..31}
                const __bf16* p = &Bs[buf][wn * 64 + nt * 16 + ln16][kk * 32 + lh * 8];
                fb[nt].q[0] = *(const uint4*)p;
                fb[nt].q[1] = *(const uint4*)(p + 16);
            }
#pragma unroll
            for (int mt = 0; mt < 2; ++mt)
#pragma unroll
                for (int nt = 0; nt < 4; ++nt)
                    acc[mt][nt] = wmma_bf16(fa[mt], fb[nt], acc[mt][nt]);
        }

        if (kt + 1 < nk) {
            sstore(buf ^ 1);
            __syncthreads();
        }
    }

#pragma unroll
    for (int mt = 0; mt < 2; ++mt)
#pragma unroll
        for (int nt = 0; nt < 4; ++nt) {
            const int gcol = bn + wn * 64 + nt * 16 + ln16;
            const float bz = bias[gcol];
#pragma unroll
            for (int r = 0; r < 8; ++r) {
                const int grow = bm + wm * 32 + mt * 16 + r + lh * 8;
                const size_t idx = (size_t)grow * N + gcol;
                float v = acc[mt][nt][r] + bz;
                if (EPI == 2) {
                    ((float*)outp)[idx] = v + resid[idx];
                } else {
                    if (EPI == 1)
                        v = 0.5f * v * (1.0f + erff(v * 0.70710678118654752f));
                    ((__bf16*)outp)[idx] = f2bf(v);
                }
            }
        }
}

// ---------------------------------------------------------------------------
// Flash attention, causal, 16 heads x head_dim 64.
// Grid: (B*H*T/64) blocks of 128 threads (4 waves); wave = 16 query rows.
// Double-buffered 32-key KV tiles in LDS; V stored transposed [d][key].
// Next tile is prefetched (WGP scope) before compute; loads issued after
// compute hit the near caches. No long-lived staging registers -> no spills.
// ---------------------------------------------------------------------------
__global__ __launch_bounds__(128)
void attn_wmma(const __bf16* __restrict__ qkv, __bf16* __restrict__ outp) {
    __shared__ __align__(16) __bf16 Kt[2][32][72];    // [key][d]
    __shared__ __align__(16) __bf16 Vt[2][64][40];    // [d][key] (transposed)
    __shared__ __align__(16) __bf16 Ps[4][16][40];    // per-wave P staging

    const int qt = blockIdx.x & 31;       // query tile (64 rows)
    const int bh = blockIdx.x >> 5;
    const int bb = bh >> 4, h = bh & 15;
    const int t = threadIdx.x;
    const int lane = t & 31, w = t >> 5;
    const int lh = lane >> 4, ln16 = lane & 15;

    const size_t rowstride = 3072;
    const size_t base = (size_t)bb * 2048 * rowstride;

    // Q fragments: rows qt*64 + w*16 + (lane%16), d split into two K=32 chunks
    FragBF qf[2];
    {
        const size_t qoff = base + (size_t)(qt * 64 + w * 16 + ln16) * rowstride + h * 64;
#pragma unroll
        for (int kk = 0; kk < 2; ++kk) {
            const __bf16* p = qkv + qoff + kk * 32 + lh * 8;
            qf[kk].q[0] = *(const uint4*)p;
            qf[kk].q[1] = *(const uint4*)(p + 16);
        }
    }

    v8f o[4];
    float m[8], l[8];
#pragma unroll
    for (int dn = 0; dn < 4; ++dn)
#pragma unroll
        for (int r = 0; r < 8; ++r) o[dn][r] = 0.0f;
#pragma unroll
    for (int r = 0; r < 8; ++r) { m[r] = -1e30f; l[r] = 0.0f; }

    const int key_t = t >> 2;          // 0..31 : key row handled by this thread
    const int dV = (t & 3) * 16;       // 0,16,32,48 : d chunk
    const int jmax = 2 * qt + 1;
    // wave-uniform causal bound as an SGPR (clean s_cbranch, EXEC untouched)
    const int myhi = __builtin_amdgcn_readfirstlane(qt * 64 + w * 16 + 15);

    auto kvcopy = [&](int jt, int buf) {   // global -> LDS, short register lives
        const size_t koff = base + (size_t)(jt * 32 + key_t) * rowstride + 1024 + h * 64 + dV;
        Pack16 rk, rv;
        rk.q[0] = ((const uint4*)(qkv + koff))[0];
        rk.q[1] = ((const uint4*)(qkv + koff))[1];
        rv.q[0] = ((const uint4*)(qkv + koff + 1024))[0];
        rv.q[1] = ((const uint4*)(qkv + koff + 1024))[1];
        *(uint4*)&Kt[buf][key_t][dV]     = rk.q[0];
        *(uint4*)&Kt[buf][key_t][dV + 8] = rk.q[1];
#pragma unroll
        for (int i = 0; i < 16; ++i) Vt[buf][dV + i][key_t] = rv.h[i];
    };

    kvcopy(0, 0);
    __syncthreads();

    for (int jt = 0; jt <= jmax; ++jt) {
        const int buf = jt & 1;
        if (jt < jmax) {  // warm WGP-level caches for next tile; zero reg cost
            const size_t pko = base + (size_t)((jt + 1) * 32 + key_t) * rowstride
                             + 1024 + h * 64 + dV;
            __builtin_prefetch(qkv + pko, 0, 3);
            __builtin_prefetch(qkv + pko + 1024, 0, 3);
        }

        if (jt * 32 <= myhi) {                // scalar-uniform: EXEC stays full
            // S = Q K^T  (two 16-key N subtiles, d accumulated over two WMMAs)
            v8f s[2];
#pragma unroll
            for (int n = 0; n < 2; ++n) {
#pragma unroll
                for (int r = 0; r < 8; ++r) s[n][r] = 0.0f;
                FragBF bk0, bk1;
                const __bf16* p0 = &Kt[buf][n * 16 + ln16][lh * 8];
                bk0.q[0] = *(const uint4*)p0;
                bk0.q[1] = *(const uint4*)(p0 + 16);
                const __bf16* p1 = p0 + 32;
                bk1.q[0] = *(const uint4*)p1;
                bk1.q[1] = *(const uint4*)(p1 + 16);
                s[n] = wmma_bf16(qf[0], bk0, s[n]);
                s[n] = wmma_bf16(qf[1], bk1, s[n]);
            }

            // scale + causal mask + online softmax, fused per row to keep
            // live ranges short (no alpha[] array, P staged immediately)
#pragma unroll
            for (int r = 0; r < 8; ++r) {
                const int qrow = qt * 64 + w * 16 + r + lh * 8;
#pragma unroll
                for (int n = 0; n < 2; ++n) {
                    const int key = jt * 32 + n * 16 + ln16;
                    const float v = s[n][r] * 0.125f;   // 1/sqrt(64)
                    s[n][r] = (key <= qrow) ? v : -1e30f;
                }
                float mx = fmaxf(s[0][r], s[1][r]);
#pragma unroll
                for (int off = 8; off; off >>= 1) mx = fmaxf(mx, __shfl_xor(mx, off));
                const float mn = fmaxf(m[r], mx);
                const float alpha = __expf(m[r] - mn);
                m[r] = mn;
                const float p0 = __expf(s[0][r] - mn);
                const float p1 = __expf(s[1][r] - mn);
                float rsum = p0 + p1;
#pragma unroll
                for (int off = 8; off; off >>= 1) rsum += __shfl_xor(rsum, off);
                l[r] = l[r] * alpha + rsum;
#pragma unroll
                for (int dn = 0; dn < 4; ++dn) o[dn][r] *= alpha;
                Ps[w][r + lh * 8][ln16]      = f2bf(p0);
                Ps[w][r + lh * 8][16 + ln16] = f2bf(p1);
            }

            // P (A-layout) from per-wave staging
            FragBF pf;
            const __bf16* pp = &Ps[w][ln16][lh * 8];
            pf.q[0] = *(const uint4*)pp;
            pf.q[1] = *(const uint4*)(pp + 16);

            // O += P V
#pragma unroll
            for (int dn = 0; dn < 4; ++dn) {
                FragBF fv;
                const __bf16* pv2 = &Vt[buf][dn * 16 + ln16][lh * 8];
                fv.q[0] = *(const uint4*)pv2;
                fv.q[1] = *(const uint4*)(pv2 + 16);
                o[dn] = wmma_bf16(pf, fv, o[dn]);
            }
        }

        if (jt < jmax) {
            kvcopy(jt + 1, buf ^ 1);   // prefetched above -> short latency
            __syncthreads();
        }
    }

    // normalize + write bf16 [B,T,C]
    const size_t obase = ((size_t)bb * 2048 + qt * 64 + w * 16) * 1024 + h * 64;
#pragma unroll
    for (int r = 0; r < 8; ++r) {
        const float inv = 1.0f / l[r];
        const int mrow = r + lh * 8;
#pragma unroll
        for (int dn = 0; dn < 4; ++dn)
            outp[obase + (size_t)mrow * 1024 + dn * 16 + ln16] = f2bf(o[dn][r] * inv);
    }
}

// ---------------------------------------------------------------------------
// Host-side orchestration
// ---------------------------------------------------------------------------
extern "C" void kernel_launch(void* const* d_in, const int* in_sizes, int n_in,
                              void* d_out, int out_size, void* d_ws, size_t ws_size,
                              hipStream_t stream) {
    const float* x     = (const float*)d_in[0];
    const float* ln1g  = (const float*)d_in[1];
    const float* ln1b  = (const float*)d_in[2];
    const float* wattn = (const float*)d_in[3];
    const float* battn = (const float*)d_in[4];
    const float* wproj = (const float*)d_in[5];
    const float* bproj = (const float*)d_in[6];
    const float* ln2g  = (const float*)d_in[7];
    const float* ln2b  = (const float*)d_in[8];
    const float* wfc   = (const float*)d_in[9];
    const float* bfc   = (const float*)d_in[10];
    const float* wfc2  = (const float*)d_in[11];
    const float* bfc2  = (const float*)d_in[12];

    const int Mtok = 2 * 2048;   // 4096 tokens
    const int D = 1024;

    char* ws = (char*)d_ws;
    size_t off = 0;
    auto alloc = [&](size_t bytes) {
        char* p = ws + off;
        off = (off + bytes + 255) & ~(size_t)255;
        return p;
    };
    __bf16* wa_bf  = (__bf16*)alloc((size_t)D * 3 * D * 2);
    __bf16* wp_bf  = (__bf16*)alloc((size_t)D * D * 2);
    __bf16* wf1_bf = (__bf16*)alloc((size_t)D * 4 * D * 2);
    __bf16* wf2_bf = (__bf16*)alloc((size_t)4 * D * D * 2);
    __bf16* ln_bf  = (__bf16*)alloc((size_t)Mtok * D * 2);       // reused ln1/ln2
    __bf16* qkv_bf = (__bf16*)alloc((size_t)Mtok * 3 * D * 2);
    __bf16* att_bf = (__bf16*)alloc((size_t)Mtok * D * 2);
    float*  x1     = (float*)alloc((size_t)Mtok * D * 4);
    __bf16* h_bf   = (__bf16*)alloc((size_t)Mtok * 4 * D * 2);
    (void)ws_size; (void)in_sizes; (void)n_in; (void)out_size;

    // weight conversions (fp32 -> bf16)
    cvt_f32_bf16<<<(D * 3 * D / 4 + 255) / 256, 256, 0, stream>>>(wattn, wa_bf, D * 3 * D / 4);
    cvt_f32_bf16<<<(D * D / 4 + 255) / 256, 256, 0, stream>>>(wproj, wp_bf, D * D / 4);
    cvt_f32_bf16<<<(D * 4 * D / 4 + 255) / 256, 256, 0, stream>>>(wfc, wf1_bf, D * 4 * D / 4);
    cvt_f32_bf16<<<(4 * D * D / 4 + 255) / 256, 256, 0, stream>>>(wfc2, wf2_bf, 4 * D * D / 4);

    // x -> LN1 -> QKV
    ln_bf16<<<Mtok, 256, 0, stream>>>(x, ln1g, ln1b, ln_bf);
    gemm_bf16_wmma<0><<<dim3(3 * D / 128, Mtok / 128), 256, 0, stream>>>(
        ln_bf, wa_bf, battn, nullptr, qkv_bf, Mtok, 3 * D, D);

    // causal flash attention
    attn_wmma<<<2 * 16 * (2048 / 64), 128, 0, stream>>>(qkv_bf, att_bf);

    // proj + residual -> x1
    gemm_bf16_wmma<2><<<dim3(D / 128, Mtok / 128), 256, 0, stream>>>(
        att_bf, wp_bf, bproj, x, x1, Mtok, D, D);

    // LN2 -> fc1 (GELU) -> fc2 + residual -> out
    ln_bf16<<<Mtok, 256, 0, stream>>>(x1, ln2g, ln2b, ln_bf);
    gemm_bf16_wmma<1><<<dim3(4 * D / 128, Mtok / 128), 256, 0, stream>>>(
        ln_bf, wf1_bf, bfc, nullptr, h_bf, Mtok, 4 * D, D);
    gemm_bf16_wmma<2><<<dim3(D / 128, Mtok / 128), 256, 0, stream>>>(
        h_bf, wf2_bf, bfc2, x1, (float*)d_out, Mtok, D, 4 * D);
}
